// Discriminator_81638738362985
// MI455X (gfx1250) — matrix operand order repaired and checked by reading, not compiled, not run on hardware
//
#include <hip/hip_runtime.h>
#include <hip/hip_bf16.h>

#define T_   256
#define B_   64
#define H_   1024
#define L_   2
#define G3_  (3 * H_)        // 3072
#define KC_  (H_ / 32)       // 32 k-chunks of 32
#define NT3_ (G3_ / 16)      // 192 column tiles over 3H
#define M_   (T_ * B_)       // 16384 rows for the big gi GEMM

typedef __bf16 bf16;
typedef __attribute__((ext_vector_type(16))) __bf16 v16bf;
typedef __attribute__((ext_vector_type(8)))  __bf16 v8bf;
typedef __attribute__((ext_vector_type(4)))  __bf16 v4bf;
typedef __attribute__((ext_vector_type(8)))  float  v8f;

__device__ __forceinline__ bf16 f2bf(float f) { return (bf16)f; }
__device__ __forceinline__ float sigmoidf_(float x) { return 1.0f / (1.0f + __expf(-x)); }

// ---------------------------------------------------------------------------
// A-fragment gather: 16-bit A 16x32 (contiguous K order, same as B pack). Lane l(0-15) holds row l,
// K = 0..15; lanes 16-31 hold row l-16, K = 16..31.
// aRow points at the lane's row (row-major bf16, stride H).
// ---------------------------------------------------------------------------
__device__ __forceinline__ v16bf load_a_frag(const bf16* __restrict__ aRow,
                                             int kc, int laneHi) {
  const int off = kc * 32 + laneHi * 16;
  v8bf lo = *reinterpret_cast<const v8bf*>(aRow + off);
  v8bf hi = *reinterpret_cast<const v8bf*>(aRow + off + 8);
  return __builtin_shufflevector(lo, hi, 0, 1, 2, 3, 4, 5, 6, 7,
                                 8, 9, 10, 11, 12, 13, 14, 15);
}

// B-fragment address in the packed weight buffer: per (n_tile, kc) a 32-lane
// x 16-element contiguous block; each lane's 16 bf16 are its column's K run.
__device__ __forceinline__ const v16bf* b_frag(const bf16* __restrict__ pk,
                                               int nt, int kc, int lane) {
  return reinterpret_cast<const v16bf*>(
      pk + (((size_t)nt * KC_ + kc) * 32 + lane) * 16);
}

// ---------------------------------------------------------------------------
// Embedding: out[t*B+b, h] = bf16(relu(emb[max(tok,0), h])), 4 elems/thread
// ---------------------------------------------------------------------------
__global__ __launch_bounds__(256) void embed_kernel(
    const int* __restrict__ tok, const float* __restrict__ emb,
    bf16* __restrict__ out) {
  size_t idx = ((size_t)blockIdx.x * 256 + threadIdx.x) * 4;
  int tb = (int)(idx / H_);
  int hh = (int)(idx % H_);
  int tk = tok[tb];
  if (tk < 0) tk = 0;
  const float* e = emb + (size_t)tk * H_ + hh;
  v4bf v;
  #pragma unroll
  for (int i = 0; i < 4; ++i) v[i] = f2bf(fmaxf(e[i], 0.0f));
  *reinterpret_cast<v4bf*>(out + idx) = v;
}

// ---------------------------------------------------------------------------
// Pack W [3H, H] fp32 row-major (gates stacked r,z,n) into bf16 WMMA
// B-fragments. 16-bit B 32x16: lanes 0-15 = col n, K=0..15; lanes 16-31 =
// col n, K=16..31 (contiguous K run per lane).
// ---------------------------------------------------------------------------
__global__ __launch_bounds__(256) void pack_w_kernel(
    const float* __restrict__ W, bf16* __restrict__ pk) {
  int tid = blockIdx.x * 256 + threadIdx.x;   // (nt, kc, lane)
  int lane = tid & 31;
  int kc = (tid >> 5) & (KC_ - 1);
  int nt = tid >> 10;
  if (nt >= NT3_) return;
  int laneHi = lane >> 4;
  int n = nt * 16 + (lane & 15);
  const float* wr = W + (size_t)n * H_ + kc * 32 + laneHi * 16;
  bf16* o = pk + (((size_t)nt * KC_ + kc) * 32 + lane) * 16;
  #pragma unroll
  for (int i = 0; i < 16; ++i) o[i] = f2bf(wr[i]);
}

// ---------------------------------------------------------------------------
// Big parallel GEMM: gi[m, n] = sum_k xin[m,k] * Wih[n,k] + bih[n]
// M = 16384, N = 3072, K = 1024.  grid (24, 1024), 8 waves/block,
// one wave per 16x16 tile, 32 WMMAs.
// ---------------------------------------------------------------------------
__global__ __launch_bounds__(256) void gi_gemm_kernel(
    const bf16* __restrict__ xin, const bf16* __restrict__ pk,
    const float* __restrict__ bih, float* __restrict__ gi) {
  const int lane = threadIdx.x & 31;
  const int laneHi = lane >> 4;
  const int nt = blockIdx.x * 8 + (threadIdx.x >> 5);  // 0..191
  const int mt = blockIdx.y;                           // 0..1023
  const bf16* aRow = xin + (size_t)(mt * 16 + (lane & 15)) * H_;
  v8f acc = {};
  #pragma unroll 4
  for (int kc = 0; kc < KC_; ++kc) {
    v16bf a = load_a_frag(aRow, kc, laneHi);
    v16bf b = *b_frag(pk, nt, kc, lane);
    acc = __builtin_amdgcn_wmma_f32_16x16x32_bf16(
        false, a, false, b, (short)0, acc, false, false);
  }
  const int n = nt * 16 + (lane & 15);
  const float bias = bih[n];
  #pragma unroll
  for (int i = 0; i < 8; ++i) {
    const int m = mt * 16 + i + laneHi * 8;
    gi[(size_t)m * G3_ + n] = acc[i] + bias;
  }
}

// ---------------------------------------------------------------------------
// One recurrent step: gh = h·Whhᵀ (3 gate tiles per wave, 96 WMMAs) fused
// with GRU gate math.  grid (8, 4), 8 waves/block; kernel boundary is the
// step-to-step synchronization; h ping-pongs between in/out buffers.
// ---------------------------------------------------------------------------
__global__ __launch_bounds__(256) void gru_step_kernel(
    const bf16* __restrict__ hB_in, const float* __restrict__ hF_in,
    const float* __restrict__ gi, const bf16* __restrict__ pkhh,
    const float* __restrict__ bhh,
    bf16* __restrict__ hB_out, float* __restrict__ hF_out,
    bf16* __restrict__ xout, int relu_out, int t) {
  const int lane = threadIdx.x & 31;
  const int laneHi = lane >> 4;
  const int jt = blockIdx.x * 8 + (threadIdx.x >> 5);  // 0..63 (H tile)
  const int bt = blockIdx.y;                           // 0..3  (B tile)
  const bf16* aRow = hB_in + (size_t)(bt * 16 + (lane & 15)) * H_;
  v8f accR = {}, accZ = {}, accN = {};
  #pragma unroll 2
  for (int kc = 0; kc < KC_; ++kc) {
    v16bf a = load_a_frag(aRow, kc, laneHi);
    v16bf bR = *b_frag(pkhh, jt, kc, lane);
    v16bf bZ = *b_frag(pkhh, 64 + jt, kc, lane);
    v16bf bN = *b_frag(pkhh, 128 + jt, kc, lane);
    accR = __builtin_amdgcn_wmma_f32_16x16x32_bf16(
        false, a, false, bR, (short)0, accR, false, false);
    accZ = __builtin_amdgcn_wmma_f32_16x16x32_bf16(
        false, a, false, bZ, (short)0, accZ, false, false);
    accN = __builtin_amdgcn_wmma_f32_16x16x32_bf16(
        false, a, false, bN, (short)0, accN, false, false);
  }
  const int j = jt * 16 + (lane & 15);
  const float bhr = bhh[j], bhz = bhh[H_ + j], bhn = bhh[2 * H_ + j];
  #pragma unroll
  for (int i = 0; i < 8; ++i) {
    const int m = bt * 16 + i + laneHi * 8;
    const size_t gbase = ((size_t)t * B_ + m) * (size_t)G3_;
    float gir = gi[gbase + j];
    float giz = gi[gbase + H_ + j];
    float gin = gi[gbase + 2 * H_ + j];
    float r = sigmoidf_(gir + accR[i] + bhr);
    float z = sigmoidf_(giz + accZ[i] + bhz);
    float n = tanhf(gin + r * (accN[i] + bhn));
    float hold = hF_in[(size_t)m * H_ + j];
    float hnew = (1.0f - z) * n + z * hold;
    hF_out[(size_t)m * H_ + j] = hnew;
    hB_out[(size_t)m * H_ + j] = f2bf(hnew);
    float o = relu_out ? fmaxf(hnew, 0.0f) : hnew;
    xout[((size_t)t * B_ + m) * H_ + j] = f2bf(o);
  }
}

// ---------------------------------------------------------------------------
// State init helpers
// ---------------------------------------------------------------------------
__global__ __launch_bounds__(256) void init_h_zero(float* __restrict__ hF,
                                                   bf16* __restrict__ hB) {
  int i = blockIdx.x * 256 + threadIdx.x;
  hF[i] = 0.0f;
  hB[i] = f2bf(0.0f);
}

__global__ __launch_bounds__(256) void copy_h(const float* __restrict__ src,
                                              float* __restrict__ hF,
                                              bf16* __restrict__ hB) {
  int i = blockIdx.x * 256 + threadIdx.x;
  float v = src[i];
  hF[i] = v;
  hB[i] = f2bf(v);
}

// ---------------------------------------------------------------------------
// Score: sigmoid(cat(cell0, cell1) · Wᵀ + b), one wave per (t, b)
// ---------------------------------------------------------------------------
__global__ __launch_bounds__(256) void score_kernel(
    const bf16* __restrict__ c0, const bf16* __restrict__ c1,
    const float* __restrict__ w, const float* __restrict__ bptr,
    float* __restrict__ out) {
  int gw = (blockIdx.x * 256 + threadIdx.x) >> 5;  // = t*B + b
  int lane = threadIdx.x & 31;
  if (gw >= M_) return;
  const bf16* r0 = c0 + (size_t)gw * H_;
  const bf16* r1 = c1 + (size_t)gw * H_;
  float s = 0.0f;
  for (int j = lane; j < H_; j += 32)
    s += (float)r0[j] * w[j] + (float)r1[j] * w[H_ + j];
  #pragma unroll
  for (int m = 16; m >= 1; m >>= 1) s += __shfl_xor(s, m, 32);
  if (lane == 0) out[gw] = sigmoidf_(s + bptr[0]);
}

// ---------------------------------------------------------------------------
extern "C" void kernel_launch(void* const* d_in, const int* in_sizes, int n_in,
                              void* d_out, int out_size, void* d_ws,
                              size_t ws_size, hipStream_t stream) {
  (void)in_sizes; (void)n_in; (void)out_size; (void)ws_size;

  const int*   tok    = (const int*)d_in[0];
  const float* emb    = (const float*)d_in[1];
  const float* encWih = (const float*)d_in[2];
  const float* encWhh = (const float*)d_in[3];
  const float* encBih = (const float*)d_in[4];
  const float* encBhh = (const float*)d_in[5];
  const float* gruWih = (const float*)d_in[6];
  const float* gruWhh = (const float*)d_in[7];
  const float* gruBih = (const float*)d_in[8];
  const float* gruBhh = (const float*)d_in[9];
  const float* celWih = (const float*)d_in[10];
  const float* celWhh = (const float*)d_in[11];
  const float* celBih = (const float*)d_in[12];
  const float* celBhh = (const float*)d_in[13];
  const float* scoreW = (const float*)d_in[14];
  const float* scoreB = (const float*)d_in[15];

  const size_t WSTRIDE = (size_t)G3_ * H_;  // per-layer weight stride (elems)

  // ---- workspace carve-out ----
  char* ws = (char*)d_ws;
  size_t off = 0;
  auto take = [&](size_t bytes) -> char* {
    char* p = ws + off;
    off += (bytes + 255) & ~(size_t)255;
    return p;
  };
  bf16*  x0   = (bf16*)take((size_t)M_ * H_ * sizeof(bf16));   // embed / cell0 out
  bf16*  actA = (bf16*)take((size_t)M_ * H_ * sizeof(bf16));   // ping / cell1 out
  bf16*  actB = (bf16*)take((size_t)M_ * H_ * sizeof(bf16));   // pong
  float* gi   = (float*)take((size_t)M_ * G3_ * sizeof(float));
  bf16*  pkih = (bf16*)take(WSTRIDE * sizeof(bf16));
  bf16*  pkhh = (bf16*)take(WSTRIDE * sizeof(bf16));
  float* hFb  = (float*)take((size_t)6 * 2 * B_ * H_ * sizeof(float));
  bf16*  hBb  = (bf16*) take((size_t)6 * 2 * B_ * H_ * sizeof(bf16));
  auto hF = [&](int p, int par) { return hFb + ((size_t)p * 2 + par) * B_ * H_; };
  auto hB = [&](int p, int par) { return hBb + ((size_t)p * 2 + par) * B_ * H_; };

  // ---- pass table: enc0, enc1, gru0, gru1, cell0, cell1 ----
  const float* Wih_p[6] = { encWih, encWih + WSTRIDE, gruWih, gruWih + WSTRIDE,
                            celWih, celWih + WSTRIDE };
  const float* Whh_p[6] = { encWhh, encWhh + WSTRIDE, gruWhh, gruWhh + WSTRIDE,
                            celWhh, celWhh + WSTRIDE };
  const float* bih_p[6] = { encBih, encBih + G3_, gruBih, gruBih + G3_,
                            celBih, celBih + G3_ };
  const float* bhh_p[6] = { encBhh, encBhh + G3_, gruBhh, gruBhh + G3_,
                            celBhh, celBhh + G3_ };
  bf16* xin_p[6]  = { x0,   actA, actB, actA, actB, actB };
  bf16* xout_p[6] = { actA, actB, actA, actB, x0,   actA };
  const int relu_p[6] = { 0, 1, 0, 0, 0, 0 };  // relu only between enc and gru
  const int h0src[6]  = { -1, -1, 0, 1, 2, 3 }; // seed state from pass's final

  const int PACK_BLOCKS = (NT3_ * KC_ * 32) / 256;      // 768
  const int HN_BLOCKS   = (B_ * H_) / 256;              // 256

  // ---- 1. embedding + relu -> bf16 ----
  embed_kernel<<<(size_t)M_ * H_ / 4 / 256, 256, 0, stream>>>(tok, emb, x0);

  // ---- 2. six sequential GRU layer passes ----
  for (int p = 0; p < 6; ++p) {
    pack_w_kernel<<<PACK_BLOCKS, 256, 0, stream>>>(Wih_p[p], pkih);
    pack_w_kernel<<<PACK_BLOCKS, 256, 0, stream>>>(Whh_p[p], pkhh);

    // parallel input-side GEMM for all T steps (fused bih)
    gi_gemm_kernel<<<dim3(NT3_ / 8, M_ / 16), 256, 0, stream>>>(
        xin_p[p], pkih, bih_p[p], gi);

    // seed hidden state
    if (h0src[p] < 0)
      init_h_zero<<<HN_BLOCKS, 256, 0, stream>>>(hF(p, 0), hB(p, 0));
    else
      copy_h<<<HN_BLOCKS, 256, 0, stream>>>(hF(h0src[p], 0), hF(p, 0), hB(p, 0));

    // sequential recurrence; kernel boundary = step sync; ping-pong state
    for (int t = 0; t < T_; ++t) {
      const int pi = t & 1, po = (t + 1) & 1;
      gru_step_kernel<<<dim3(H_ / 128, B_ / 16), 256, 0, stream>>>(
          hB(p, pi), hF(p, pi), gi, pkhh, bhh_p[p],
          hB(p, po), hF(p, po), xout_p[p], relu_p[p], t);
    }
  }

  // ---- 3. score head ----
  score_kernel<<<(M_ * 32) / 256, 256, 0, stream>>>(
      x0 /*cell0 out*/, actA /*cell1 out*/, scoreW, scoreB, (float*)d_out);
}
